// FlashMLA_76536317215203
// MI455X (gfx1250) — matrix operand order repaired and checked by reading, not compiled, not run on hardware
//
#include <hip/hip_runtime.h>
#include <hip/hip_bf16.h>

// ---------------------------------------------------------------------------
// MLA forward for MI455X (gfx1250, wave32, WMMA 16x16x32 bf16)
// B=2, S=2048, H=2048, NH=16, HD=128, L=512
// ---------------------------------------------------------------------------

#ifndef USE_ASYNC_COPY
#define USE_ASYNC_COPY 1   // gfx1250 GLOBAL_LOAD_ASYNC_TO_LDS_B128 path
#endif

typedef unsigned short u16;
typedef unsigned int   u32;
typedef __attribute__((ext_vector_type(16))) __bf16          v16bf;
typedef __attribute__((ext_vector_type(16))) unsigned short  v16us;
typedef __attribute__((ext_vector_type(8)))  float           v8f;

__device__ __forceinline__ u16 f2bf(float f) {
    u32 x = __builtin_bit_cast(u32, f);
    u32 r = x + 0x7FFFu + ((x >> 16) & 1u);   // round-to-nearest-even
    return (u16)(r >> 16);
}

__device__ __forceinline__ v8f wmma_bf16(v16bf a, v16bf b, v8f c) {
    return __builtin_amdgcn_wmma_f32_16x16x32_bf16(
        false, a, false, b, (short)0, c, false, false);
}

// 16-byte global -> LDS copy. Async (ASYNCcnt) on the CDNA5 path.
__device__ __forceinline__ void cp_b128(u16* ldst, const u16* gsrc) {
#if USE_ASYNC_COPY
    unsigned lds_off = (unsigned)(unsigned long long)ldst;          // low 32b = LDS addr
    unsigned long long ga = (unsigned long long)gsrc;
    asm volatile("global_load_async_to_lds_b128 %0, %1, off"
                 :: "v"(lds_off), "v"(ga) : "memory");
#else
    *(uint4*)ldst = *(const uint4*)gsrc;
#endif
}
__device__ __forceinline__ void cp_wait() {
#if USE_ASYNC_COPY
    asm volatile("s_wait_asynccnt 0x0" ::: "memory");
#endif
}

// load a 16x32 bf16 A-fragment (two contiguous 16B halves) straight from global
__device__ __forceinline__ v16bf ld_afrag_global(const u16* row_ptr) {
    union { v16us v; uint4 q[2]; } u;
    u.q[0] = *(const uint4*)(row_ptr);
    u.q[1] = *(const uint4*)(row_ptr + 16);
    return __builtin_bit_cast(v16bf, u.v);
}

// ---------------------------------------------------------------------------
// fp32 -> bf16 conversion (grid-stride)
// ---------------------------------------------------------------------------
__global__ void cvt_bf16_kernel(const float* __restrict__ in, u16* __restrict__ out, int n) {
    int i = blockIdx.x * blockDim.x + threadIdx.x;
    int stride = gridDim.x * blockDim.x;
    for (; i < n; i += stride) out[i] = f2bf(in[i]);
}

// ---------------------------------------------------------------------------
// bf16 GEMM: C[M,N] = A[M,K] @ W[N,K]^T (both row-major).
// 128 threads = 4 waves (2x2), per-wave 64x64 = 4x4 accumulators, K-step 32.
// Double-buffered LDS, async global->LDS staging, batched fragment loads.
// MODE 0: scatter bf16 to q-heads [B,NH,S,HD]
// MODE 1: cols<2048 -> K heads [B,NH,S,HD]; cols>=2048 -> V transposed [B,NH,HD,S]
// MODE 2: fp32 row-major output
// ---------------------------------------------------------------------------
#define GTM 128
#define GTN 128
#define GTK 32
#define GSTR 40   // LDS row stride (u16) = 80 bytes

template<int MODE>
__global__ __launch_bounds__(128)
void gemm_bf16_kernel(const u16* __restrict__ A, const u16* __restrict__ Bw,
                      void* __restrict__ out0, void* __restrict__ out1,
                      int M, int N, int K)
{
    __shared__ __attribute__((aligned(16))) u16 Als[2][GTM][GSTR];
    __shared__ __attribute__((aligned(16))) u16 Bls[2][GTN][GSTR];

    const int tid  = threadIdx.x;
    const int wave = tid >> 5;
    const int lane = tid & 31;
    const int lo   = lane & 15;
    const int hi   = lane >> 4;
    const int wm   = wave & 1;    // 0..1 (M)
    const int wn   = wave >> 1;   // 0..1 (N)
    const int m0   = blockIdx.y * GTM;
    const int n0   = blockIdx.x * GTN;

    v8f acc[4][4];
#pragma unroll
    for (int i = 0; i < 4; ++i)
#pragma unroll
        for (int j = 0; j < 4; ++j)
#pragma unroll
            for (int e = 0; e < 8; ++e) acc[i][j][e] = 0.0f;

    const int NK = K / GTK;

    // issue tile 0 into buffer 0 (4 b128 per matrix per thread)
#pragma unroll
    for (int i = 0; i < 4; ++i) {
        int lin = (tid + i * 128) * 8;       // 0..4088
        int r = lin >> 5, c = lin & 31;
        cp_b128(&Als[0][r][c], A  + (size_t)(m0 + r) * K + c);
        cp_b128(&Bls[0][r][c], Bw + (size_t)(n0 + r) * K + c);
    }

    for (int kt = 0; kt < NK; ++kt) {
        const int cur = kt & 1;
        cp_wait();               // copies for buffer `cur` complete
        __syncthreads();         // visible to all waves; prev reads of cur^1 done

        if (kt + 1 < NK) {
            const int k1 = (kt + 1) * GTK;
#pragma unroll
            for (int i = 0; i < 4; ++i) {
                int lin = (tid + i * 128) * 8;
                int r = lin >> 5, c = lin & 31;
                cp_b128(&Als[cur ^ 1][r][c], A  + (size_t)(m0 + r) * K + k1 + c);
                cp_b128(&Bls[cur ^ 1][r][c], Bw + (size_t)(n0 + r) * K + k1 + c);
            }
        }
        if (kt + 2 < NK) {       // L2 prefetch two tiles ahead
            const int k2 = (kt + 2) * GTK;
            int r = tid & 127;
            __builtin_prefetch(A  + (size_t)(m0 + r) * K + k2, 0, 1);
            __builtin_prefetch(Bw + (size_t)(n0 + r) * K + k2, 0, 1);
        }

        // batched fragment loads from buffer `cur`
        v16bf af[4], bf[4];
#pragma unroll
        for (int i = 0; i < 4; ++i) {
            int row = wm * 64 + i * 16 + lo;
            int kb  = hi * 8;
            v16us t;
#pragma unroll
            for (int j = 0; j < 8; ++j) {
                t[j]     = Als[cur][row][kb + j];
                t[8 + j] = Als[cur][row][16 + kb + j];
            }
            af[i] = __builtin_bit_cast(v16bf, t);
        }
#pragma unroll
        for (int i = 0; i < 4; ++i) {
            int col = wn * 64 + i * 16 + lo;
            int kb  = hi * 8;
            v16us t;
#pragma unroll
            for (int j = 0; j < 16; ++j) t[j] = Bls[cur][col][kb + (j & 7) + (j >> 3) * 16];
            bf[i] = __builtin_bit_cast(v16bf, t);
        }
#pragma unroll
        for (int i = 0; i < 4; ++i)
#pragma unroll
            for (int j = 0; j < 4; ++j)
                acc[i][j] = wmma_bf16(af[i], bf[j], acc[i][j]);
    }

    // epilogue: element e maps to m_local = hi*8+e, n_local = lo
#pragma unroll
    for (int i = 0; i < 4; ++i) {
#pragma unroll
        for (int jn = 0; jn < 4; ++jn) {
#pragma unroll
            for (int e = 0; e < 8; ++e) {
                int m = m0 + wm * 64 + i * 16 + hi * 8 + e;
                int n = n0 + wn * 64 + jn * 16 + lo;
                float v = acc[i][jn][e];
                if (MODE == 0) {
                    int bb = m >> 11, ss = m & 2047;
                    int hh = n >> 7,  dd = n & 127;
                    ((u16*)out0)[(((size_t)(bb * 16 + hh) * 2048) + ss) * 128 + dd] = f2bf(v);
                } else if (MODE == 1) {
                    int bb = m >> 11, ss = m & 2047;
                    if (n < 2048) {
                        int hh = n >> 7, dd = n & 127;
                        ((u16*)out0)[(((size_t)(bb * 16 + hh) * 2048) + ss) * 128 + dd] = f2bf(v);
                    } else {
                        int c2 = n - 2048;
                        int hh = c2 >> 7, dd = c2 & 127;
                        // V stored transposed: [B,NH,HD,S]
                        ((u16*)out1)[(((size_t)(bb * 16 + hh) * 128 + dd) * 2048) + ss] = f2bf(v);
                    }
                } else {
                    ((float*)out0)[(size_t)m * N + n] = v;
                }
            }
        }
    }
}

// ---------------------------------------------------------------------------
// Flash attention (causal), per (b, h, 128-row q block), 256 threads = 8 waves.
// Q fragments live in registers (loaded once, straight from global);
// K / V^T tiles staged to LDS with async copies; P round-trips through LDS.
// Qh/Kh: bf16 [B,NH,S,HD]; Vtg: bf16 [B,NH,HD,S]; Ao: bf16 [B*S, H]
// ---------------------------------------------------------------------------
#define ABQ 128
#define ABK 64

__global__ __launch_bounds__(256)
void mla_attn_kernel(const u16* __restrict__ Qh, const u16* __restrict__ Kh,
                     const u16* __restrict__ Vtg, u16* __restrict__ Ao)
{
    __shared__ __attribute__((aligned(16))) u16 Kls[64][136];   // [key][d]
    __shared__ __attribute__((aligned(16))) u16 Vt[128][72];    // [d][key]
    __shared__ __attribute__((aligned(16))) u16 Pls[8][16][72];

    const int tid  = threadIdx.x;
    const int wave = tid >> 5;     // 0..7, owns q rows [q0+wave*16, +16)
    const int lane = tid & 31;
    const int lo   = lane & 15;
    const int hi   = lane >> 4;
    const int q0   = blockIdx.x * ABQ;
    const int h    = blockIdx.y;
    const int b    = blockIdx.z;
    const size_t base = (size_t)(b * 16 + h) * 2048 * 128;   // plane size for Q/K/Vt
    const float scale = 0.08838834764831845f;                 // 1/sqrt(128)

    // Q A-fragments: loop-invariant, straight from global into registers.
    // lane holds row = q0 + wave*16 + lo; chunk ks covers d = [ks*32, ks*32+32)
    v16bf qf[4];
    {
        const u16* qrowp = Qh + base + (size_t)(q0 + wave * 16 + lo) * 128 + hi * 8;
#pragma unroll
        for (int ks = 0; ks < 4; ++ks) qf[ks] = ld_afrag_global(qrowp + ks * 32);
    }

    float mrow[8], lrow[8];
    v8f oacc[8];
#pragma unroll
    for (int j = 0; j < 8; ++j) { mrow[j] = -1e30f; lrow[j] = 0.0f; }
#pragma unroll
    for (int nt = 0; nt < 8; ++nt)
#pragma unroll
        for (int e = 0; e < 8; ++e) oacc[nt][e] = 0.0f;

    const int nkb = q0 / ABK + 2;   // causal: key blocks cover keys <= q0+127
    for (int kbi = 0; kbi < nkb; ++kbi) {
        const int kb = kbi * ABK;
        __syncthreads();            // all waves done reading previous K/V tiles
        // stage K tile (64 keys x 128 d) and Vt tile (128 d x 64 keys), async
#pragma unroll
        for (int i = 0; i < 4; ++i) {
            int lin = (tid + i * 256) * 8;          // 0..8184
            int rk = lin >> 7, ck = lin & 127;
            cp_b128(&Kls[rk][ck], Kh + base + (size_t)(kb + rk) * 128 + ck);
            int rv = lin >> 6, cv = lin & 63;
            cp_b128(&Vt[rv][cv], Vtg + base + (size_t)rv * 2048 + kb + cv);
        }
        cp_wait();
        __syncthreads();

        // S = Q K^T for this wave's 16 rows x 64 key cols (batched frags)
        v8f sacc[4];
#pragma unroll
        for (int nt = 0; nt < 4; ++nt)
#pragma unroll
            for (int e = 0; e < 8; ++e) sacc[nt][e] = 0.0f;

#pragma unroll
        for (int ks = 0; ks < 4; ++ks) {               // d in chunks of 32
            v16bf bf[4];
#pragma unroll
            for (int nt = 0; nt < 4; ++nt) {
                int col = nt * 16 + lo;                // key within block
                int kbv = hi * 8;
                v16us t;
#pragma unroll
                for (int j = 0; j < 16; ++j) t[j] = Kls[col][ks * 32 + kbv + (j & 7) + (j >> 3) * 16];
                bf[nt] = __builtin_bit_cast(v16bf, t);
            }
#pragma unroll
            for (int nt = 0; nt < 4; ++nt)
                sacc[nt] = wmma_bf16(qf[ks], bf[nt], sacc[nt]);
        }

        // online softmax per row (row = q0 + wave*16 + hi*8 + j)
#pragma unroll
        for (int j = 0; j < 8; ++j) {
            int qrow = q0 + wave * 16 + hi * 8 + j;
            float sv[4];
            float mx = -1e30f;
#pragma unroll
            for (int nt = 0; nt < 4; ++nt) {
                int kcol = kb + nt * 16 + lo;
                float s = sacc[nt][j] * scale;
                if (kcol > qrow) s = -1e30f;           // causal mask
                sv[nt] = s;
                mx = fmaxf(mx, s);
            }
#pragma unroll
            for (int d = 1; d < 16; d <<= 1) mx = fmaxf(mx, __shfl_xor(mx, d, 32));
            float mnew  = fmaxf(mrow[j], mx);
            float alpha = __expf(mrow[j] - mnew);
            float rs = 0.0f;
#pragma unroll
            for (int nt = 0; nt < 4; ++nt) {
                float p = __expf(sv[nt] - mnew);
                rs += p;
                Pls[wave][hi * 8 + j][nt * 16 + lo] = f2bf(p);
            }
#pragma unroll
            for (int d = 1; d < 16; d <<= 1) rs += __shfl_xor(rs, d, 32);
            lrow[j] = lrow[j] * alpha + rs;
            mrow[j] = mnew;
#pragma unroll
            for (int nt = 0; nt < 8; ++nt) oacc[nt][j] *= alpha;
        }

        // O += P @ V  (wave-local LDS round trip; per-wave LDS ops are in-order)
#pragma unroll
        for (int ks = 0; ks < 2; ++ks) {               // keys in chunks of 32
            v16bf af;
            {
                int row = lo;
                int kbv = hi * 8;
                v16us t;
#pragma unroll
                for (int j = 0; j < 8; ++j) {
                    t[j]     = Pls[wave][row][ks * 32 + kbv + j];
                    t[8 + j] = Pls[wave][row][ks * 32 + 16 + kbv + j];
                }
                af = __builtin_bit_cast(v16bf, t);
            }
            v16bf bf[8];
#pragma unroll
            for (int nt = 0; nt < 8; ++nt) {
                int col = nt * 16 + lo;                // output d
                int kbv = hi * 8;
                v16us t;
#pragma unroll
                for (int j = 0; j < 16; ++j) t[j] = Vt[col][ks * 32 + kbv + (j & 7) + (j >> 3) * 16];
                bf[nt] = __builtin_bit_cast(v16bf, t);
            }
#pragma unroll
            for (int nt = 0; nt < 8; ++nt)
                oacc[nt] = wmma_bf16(af, bf[nt], oacc[nt]);
        }
    }

    // normalize and write merged-head bf16 output [B*S, 2048]
#pragma unroll
    for (int nt = 0; nt < 8; ++nt) {
#pragma unroll
        for (int j = 0; j < 8; ++j) {
            int qrow = q0 + wave * 16 + hi * 8 + j;
            int dd   = nt * 16 + lo;
            float v  = oacc[nt][j] / lrow[j];
            Ao[(size_t)(b * 2048 + qrow) * 2048 + h * 128 + dd] = f2bf(v);
        }
    }
}

// ---------------------------------------------------------------------------
// Host-side orchestration
// ---------------------------------------------------------------------------
extern "C" void kernel_launch(void* const* d_in, const int* in_sizes, int n_in,
                              void* d_out, int out_size, void* d_ws, size_t ws_size,
                              hipStream_t stream)
{
    const float* q    = (const float*)d_in[0];   // [2,2048,2048]
    const float* lat  = (const float*)d_in[1];   // [2,2048,512]
    const float* Wq   = (const float*)d_in[2];   // [2048,2048]
    const float* Wkv  = (const float*)d_in[3];   // [4096,512]
    const float* Wout = (const float*)d_in[4];   // [2048,2048]
    float* out = (float*)d_out;                  // [2,2048,2048] fp32

    // workspace layout (bf16 intermediates, ~104 MiB — fits in 192 MB L2)
    u16* p      = (u16*)d_ws;
    u16* qbf    = p; p += 8388608;   // [4096,2048]
    u16* latbf  = p; p += 2097152;   // [4096,512]
    u16* wqbf   = p; p += 4194304;   // [2048,2048]
    u16* wkvbf  = p; p += 2097152;   // [4096,512]
    u16* wobf   = p; p += 4194304;   // [2048,2048]
    u16* qh     = p; p += 8388608;   // [B,NH,S,HD]
    u16* kh     = p; p += 8388608;   // [B,NH,S,HD]
    u16* vt     = p; p += 8388608;   // [B,NH,HD,S]  (V transposed)
    u16* ao     = p; p += 8388608;   // [4096,2048]

    // 1) fp32 -> bf16
    cvt_bf16_kernel<<<2048, 256, 0, stream>>>(q,    qbf,   8388608);
    cvt_bf16_kernel<<<1024, 256, 0, stream>>>(lat,  latbf, 2097152);
    cvt_bf16_kernel<<<1024, 256, 0, stream>>>(Wq,   wqbf,  4194304);
    cvt_bf16_kernel<<<1024, 256, 0, stream>>>(Wkv,  wkvbf, 2097152);
    cvt_bf16_kernel<<<1024, 256, 0, stream>>>(Wout, wobf,  4194304);

    // 2) qp = q @ Wq^T  -> head-major bf16
    gemm_bf16_kernel<0><<<dim3(2048 / GTN, 4096 / GTM), 128, 0, stream>>>(
        qbf, wqbf, qh, nullptr, 4096, 2048, 2048);

    // 3) kv = latent @ Wkv^T -> K head-major, V head-transposed bf16
    gemm_bf16_kernel<1><<<dim3(4096 / GTN, 4096 / GTM), 128, 0, stream>>>(
        latbf, wkvbf, kh, vt, 4096, 4096, 512);

    // 4) causal flash attention per (b, h, q-block)
    mla_attn_kernel<<<dim3(2048 / ABQ, 16, 2), 256, 0, stream>>>(qh, kh, vt, ao);

    // 5) out = attn @ Wout^T -> fp32
    gemm_bf16_kernel<2><<<dim3(2048 / GTN, 4096 / GTM), 128, 0, stream>>>(
        ao, wobf, out, nullptr, 4096, 2048, 2048);
}
